// SelfAttention_89464168775736
// MI455X (gfx1250) — matrix operand order repaired and checked by reading, not compiled, not run on hardware
//
#include <hip/hip_runtime.h>
#include <hip/hip_bf16.h>

// ---------------------------------------------------------------------------
// Self-attention forward for MI455X (gfx1250), f16 WMMA with fp32 accumulate.
// B=4, T=2048, C=1024, H=16, D=64.
// Pipeline: [cvt x,w -> f16] -> qkv GEMM (LDS-staged) ->
//           causal flash attention (block-shared K/V tiles in LDS) ->
//           out GEMM (LDS-staged)
// ---------------------------------------------------------------------------

#define BB 4
#define TT 2048
#define CC 1024
#define HH 16
#define DD 64

#define LDT 48                                   // GEMM LDS tile stride (halves)
#define LDK 72                                   // K-tile stride (halves)
#define LDV 48                                   // V-tile stride (halves)

typedef __attribute__((ext_vector_type(16))) _Float16 v16h;
typedef __attribute__((ext_vector_type(8)))  _Float16 v8h;
typedef __attribute__((ext_vector_type(8)))  float    v8f;

// D = A(16x32 f16) * B(32x16 f16) + C(16x16 f32)
__device__ __forceinline__ v8f wmma16(v16h a, v16h b, v8f c) {
    return __builtin_amdgcn_wmma_f32_16x16x32_f16(
        /*neg_a=*/false, a, /*neg_b=*/false, b,
        /*c_mod=*/(short)0, c, /*reuse_a=*/false, /*reuse_b=*/false);
}

// ---- A fragment (16x32, f16): lane L<16 -> row L, K = 0..7 & 16..23;
//      lane L+16 -> row L, K = 8..15 & 24..31.  (ISA 7.12.2 16-bit A table)
__device__ __forceinline__ v16h afrag_f16(const _Float16* base, int ld) {
    const int lane = threadIdx.x & 31;
    const int row  = lane & 15;
    const int kb   = (lane >> 4) << 3;           // 0 or 8
    const _Float16* p = base + row * ld + kb;
    v8h lo = *(const v8h*)(p);
    v8h hi = *(const v8h*)(p + 16);
    v16h r;
#pragma unroll
    for (int i = 0; i < 8; ++i) { r[i] = lo[i]; r[8 + i] = hi[i]; }
    return r;
}

// ---- B fragment (32x16, f16): lane L<16 -> col L, K = 0..7 & 16..23;
//      lane L+16 -> col L, K = 8..15 & 24..31.   (matches A fragment k order)
//      "base" points at row-major [N, K] tile origin (e.g. W rows).
__device__ __forceinline__ v16h bfrag_f16(const _Float16* base, int ld) {
    const int lane = threadIdx.x & 31;
    const int row  = lane & 15;                  // output column n
    const int kb   = (lane >> 4) << 3;           // 0 or 8
    const _Float16* p = base + row * ld + kb;
    v8h lo = *(const v8h*)(p);
    v8h hi = *(const v8h*)(p + 16);
    v16h r;
#pragma unroll
    for (int i = 0; i < 8; ++i) { r[i] = lo[i]; r[8 + i] = hi[i]; }
    return r;
}

// ---------------------------------------------------------------------------
// Kernel 0: one-shot fp32 -> f16 conversion (8 elems / thread, vectorized).
// ---------------------------------------------------------------------------
__global__ __launch_bounds__(256) void cvt_f16(
    const float* __restrict__ src, _Float16* __restrict__ dst, int n8)
{
    const int i = blockIdx.x * blockDim.x + threadIdx.x;
    if (i >= n8) return;
    const float4* s = (const float4*)src;
    const float4 a = s[2 * i], b = s[2 * i + 1];
    v8h h;
    h[0] = (_Float16)a.x; h[1] = (_Float16)a.y;
    h[2] = (_Float16)a.z; h[3] = (_Float16)a.w;
    h[4] = (_Float16)b.x; h[5] = (_Float16)b.y;
    h[6] = (_Float16)b.z; h[7] = (_Float16)b.w;
    *(v8h*)(dst + 8 * (size_t)i) = h;
}

// ---------------------------------------------------------------------------
// Shared GEMM mainloop: C[128x128] tile, LDS-staged operands, 8 waves,
// wave tile 64(M) x 32(N) = 4x2 WMMA accumulators. K stepped by 32 with a
// register prefetch of the next K-slice overlapping the WMMAs.
// ---------------------------------------------------------------------------
__device__ __forceinline__ void gemm_tile(
    const _Float16* __restrict__ A, const _Float16* __restrict__ Bw,
    _Float16* sA, _Float16* sB, int m0b, int n0b, int wm, int wn,
    v8f (&acc)[4][2])
{
    const int tid = threadIdx.x;
    int row[2], c8[2];
#pragma unroll
    for (int t = 0; t < 2; ++t) {
        const int c = tid + t * 256;             // 512 16B-chunks per tile
        row[t] = c >> 2;                         // 0..127
        c8[t]  = (c & 3) * 8;                    // half offset within row
    }

    v8h ra[2], rb[2];
#pragma unroll
    for (int t = 0; t < 2; ++t) {
        ra[t] = *(const v8h*)(A  + (size_t)(m0b + row[t]) * CC + c8[t]);
        rb[t] = *(const v8h*)(Bw + (size_t)(n0b + row[t]) * CC + c8[t]);
    }

    for (int k = 0; k < CC; k += 32) {
#pragma unroll
        for (int t = 0; t < 2; ++t) {            // regs -> LDS tile
            *(v8h*)(sA + row[t] * LDT + c8[t]) = ra[t];
            *(v8h*)(sB + row[t] * LDT + c8[t]) = rb[t];
        }
        __syncthreads();                         // tile visible to all waves

        if (k + 32 < CC) {                       // global prefetch of next slice
#pragma unroll
            for (int t = 0; t < 2; ++t) {
                ra[t] = *(const v8h*)(A  + (size_t)(m0b + row[t]) * CC + (k + 32) + c8[t]);
                rb[t] = *(const v8h*)(Bw + (size_t)(n0b + row[t]) * CC + (k + 32) + c8[t]);
            }
            if (k + 64 < CC) {
                __builtin_prefetch(A  + (size_t)m0b * CC + k + 64, 0, 3);
                __builtin_prefetch(Bw + (size_t)n0b * CC + k + 64, 0, 3);
            }
        }

        v16h a[4], bf[2];
#pragma unroll
        for (int i = 0; i < 4; ++i)
            a[i] = afrag_f16(sA + (wm * 64 + 16 * i) * LDT, LDT);
#pragma unroll
        for (int j = 0; j < 2; ++j)
            bf[j] = bfrag_f16(sB + (wn * 32 + 16 * j) * LDT, LDT);
#pragma unroll
        for (int i = 0; i < 4; ++i)
#pragma unroll
            for (int j = 0; j < 2; ++j)
                acc[i][j] = wmma16(a[i], bf[j], acc[i][j]);
        __syncthreads();                         // reads done before overwrite
    }
}

// ---------------------------------------------------------------------------
// Kernel 1: qkv = x @ qkv_w^T + qkv_b, scattered into Q[B,H,T,D], K[B,H,T,D],
// Vt[B,H,D,T] as f16.  grid = (3072/128, 8192/128) = (24, 64)
// ---------------------------------------------------------------------------
__global__ __launch_bounds__(256) void qkv_gemm(
    const _Float16* __restrict__ x, const _Float16* __restrict__ w,
    const float* __restrict__ bias,
    _Float16* __restrict__ Qb, _Float16* __restrict__ Kb,
    _Float16* __restrict__ Vt)
{
    __shared__ _Float16 sA[128 * LDT];
    __shared__ _Float16 sB[128 * LDT];
    const int wave = threadIdx.x >> 5;
    const int lane = threadIdx.x & 31;
    const int wm = wave >> 2, wn = wave & 3;
    const int m0b = blockIdx.y * 128, n0b = blockIdx.x * 128;

    v8f acc[4][2] = {};
    gemm_tile(x, w, sA, sB, m0b, n0b, wm, wn, acc);

    const int m0 = m0b + wm * 64, n0 = n0b + wn * 32;
    const int col = lane & 15;
    const int rhi = (lane >> 4) << 3;            // 0 or 8
#pragma unroll
    for (int i = 0; i < 4; ++i) {
#pragma unroll
        for (int j = 0; j < 2; ++j) {
            const int f  = n0 + 16 * j + col;    // output feature 0..3071
            const float bv = bias[f];
            const int sect = f >> 10;            // 0=q 1=k 2=v
            const int fl = f & (CC - 1);
            const int h = fl >> 6, d = fl & 63;
#pragma unroll
            for (int r = 0; r < 8; ++r) {
                const int m = m0 + 16 * i + rhi + r;   // 0..8191
                const int b = m >> 11, t = m & (TT - 1);
                const _Float16 hv = (_Float16)(acc[i][j][r] + bv);
                const size_t bh = (size_t)b * HH + h;
                if (sect == 0)
                    Qb[(bh * TT + t) * DD + d] = hv;
                else if (sect == 1)
                    Kb[(bh * TT + t) * DD + d] = hv;
                else
                    Vt[(bh * DD + d) * TT + t] = hv;
            }
        }
    }
}

// ---------------------------------------------------------------------------
// Kernel 2: causal flash attention with block-shared K/V tiles.
// Block = 8 waves = 128 query rows of one (b,h); each 32-key step stages
// K (32x64) and V^T (64x32) in LDS once, consumed by all waves.
// grid = (T/128, B*H) = (16, 64).
// ---------------------------------------------------------------------------
__global__ __launch_bounds__(256) void flash_attn(
    const _Float16* __restrict__ Qb, const _Float16* __restrict__ Kb,
    const _Float16* __restrict__ Vt, _Float16* __restrict__ O)
{
    __shared__ _Float16 sK[32 * LDK];            // keys x d   (row-major)
    __shared__ _Float16 sV[64 * LDV];            // d x keys   (row-major, = V^T)
    __shared__ _Float16 pbuf[8][16 * 32];        // per-wave P tile

    const int tid = threadIdx.x;
    const int wave = tid >> 5;
    const int lane = tid & 31;
    const int bh = blockIdx.y;
    const int b = bh >> 4, h = bh & 15;
    const int Q0 = blockIdx.x * 128;
    const int q0 = Q0 + wave * 16;

    const _Float16* Qh = Qb + (size_t)bh * TT * DD;
    const _Float16* Kh = Kb + (size_t)bh * TT * DD;
    const _Float16* Vh = Vt + (size_t)bh * DD * TT;

    const v16h qa0 = afrag_f16(Qh + (size_t)q0 * DD + 0, DD);   // d = 0..31
    const v16h qa1 = afrag_f16(Qh + (size_t)q0 * DD + 32, DD);  // d = 32..63

    v8f o[4] = {};                               // 16x64 fp32 out
    float mrow[8], lrow[8];
#pragma unroll
    for (int r = 0; r < 8; ++r) { mrow[r] = -__builtin_inff(); lrow[r] = 0.f; }

    const float scale = 0.125f;                  // 1/sqrt(64)
    const int rhi = (lane >> 4) << 3;
    const int colk = lane & 15;
    const int kend = q0 + 16;                    // this wave's causal bound
    const int KMAX = Q0 + 128;                   // block's causal bound

    // cooperative-copy indices (uniform across waves)
    const int krow = tid >> 2, kc8 = (tid & 3) * 8;        // K: 32x64 = 128 chunks*? (use 128 thr)
    const int vrow = tid >> 1, vc8 = (tid & 1) * 8;        // V: 64x32 = 128 chunks

    for (int j = 0; j < KMAX; j += 32) {
        __syncthreads();                         // prev-iter reads complete
        if (tid < 128) {                         // K tile: 32 rows x 64 halves
            *(v8h*)(sK + krow * LDK + kc8) =
                *(const v8h*)(Kh + (size_t)(j + krow) * DD + kc8);
            *(v8h*)(sK + krow * LDK + kc8 + 32) =
                *(const v8h*)(Kh + (size_t)(j + krow) * DD + kc8 + 32);
        } else {                                 // V tile: 64 rows x 32 halves
            *(v8h*)(sV + vrow * LDV + vc8 - (64 * LDV / 2) * 0) = // vrow in 64..127
                *(const v8h*)(Vh + (size_t)(vrow - 64) * TT + j + vc8);
        }
        __syncthreads();                         // tiles ready

        if (j < kend) {                          // wave-uniform causal skip
            // ---- scores from LDS K tile (two 16-key sub-tiles, D split)
            v8f s0 = {}, s1 = {};
            s0 = wmma16(qa0, bfrag_f16(sK + 0, LDK), s0);
            s0 = wmma16(qa1, bfrag_f16(sK + 32, LDK), s0);
            s1 = wmma16(qa0, bfrag_f16(sK + 16 * LDK + 0, LDK), s1);
            s1 = wmma16(qa1, bfrag_f16(sK + 16 * LDK + 32, LDK), s1);

            const int kk = j + colk;
#pragma unroll
            for (int r = 0; r < 8; ++r) {
                const int qi = q0 + rhi + r;
                s0[r] = (kk <= qi) ? s0[r] * scale : -__builtin_inff();
                s1[r] = (kk + 16 <= qi) ? s1[r] * scale : -__builtin_inff();

                float v = fmaxf(s0[r], s1[r]);
                v = fmaxf(v, __shfl_xor(v, 1, 32));
                v = fmaxf(v, __shfl_xor(v, 2, 32));
                v = fmaxf(v, __shfl_xor(v, 4, 32));
                v = fmaxf(v, __shfl_xor(v, 8, 32));          // row max
                const float mnew = fmaxf(mrow[r], v);
                const float alpha = __expf(mrow[r] - mnew);
                mrow[r] = mnew;

                const float p0 = __expf(s0[r] - mnew);
                const float p1 = __expf(s1[r] - mnew);
                float ps = p0 + p1;
                ps += __shfl_xor(ps, 1, 32);
                ps += __shfl_xor(ps, 2, 32);
                ps += __shfl_xor(ps, 4, 32);
                ps += __shfl_xor(ps, 8, 32);
                lrow[r] = lrow[r] * alpha + ps;

                o[0][r] *= alpha; o[1][r] *= alpha;
                o[2][r] *= alpha; o[3][r] *= alpha;
                s0[r] = p0; s1[r] = p1;
            }

            // ---- C-layout probs -> LDS (row-major 16x32) -> A fragment
            _Float16* pb = &pbuf[wave][0];
#pragma unroll
            for (int r = 0; r < 8; ++r) {
                const int rw = rhi + r;
                pb[rw * 32 + colk]      = (_Float16)s0[r];
                pb[rw * 32 + 16 + colk] = (_Float16)s1[r];
            }
            const v16h pf = afrag_f16(pb, 32);   // same-wave DS RAW

            // ---- O += P(16x32) * V(32x64) from shared V^T tile
#pragma unroll
            for (int dt = 0; dt < 4; ++dt) {
                const v16h vf = bfrag_f16(sV + (dt * 16) * LDV, LDV);
                o[dt] = wmma16(pf, vf, o[dt]);
            }
        }
    }

    // ---- normalize and store O[b, t, h*64+d] as f16
#pragma unroll
    for (int dt = 0; dt < 4; ++dt) {
#pragma unroll
        for (int r = 0; r < 8; ++r) {
            const int t = q0 + rhi + r;
            const int c = h * DD + dt * 16 + colk;
            O[((size_t)b * TT + t) * CC + c] = (_Float16)(o[dt][r] / lrow[r]);
        }
    }
}

// ---------------------------------------------------------------------------
// Kernel 3: y = O @ out_w^T + out_b  (f32 output). grid = (1024/128, 64).
// ---------------------------------------------------------------------------
__global__ __launch_bounds__(256) void out_gemm(
    const _Float16* __restrict__ Obuf, const _Float16* __restrict__ w,
    const float* __restrict__ bias, float* __restrict__ y)
{
    __shared__ _Float16 sA[128 * LDT];
    __shared__ _Float16 sB[128 * LDT];
    const int wave = threadIdx.x >> 5;
    const int lane = threadIdx.x & 31;
    const int wm = wave >> 2, wn = wave & 3;
    const int m0b = blockIdx.y * 128, n0b = blockIdx.x * 128;

    v8f acc[4][2] = {};
    gemm_tile(Obuf, w, sA, sB, m0b, n0b, wm, wn, acc);

    const int m0 = m0b + wm * 64, n0 = n0b + wn * 32;
    const int col = lane & 15;
    const int rhi = (lane >> 4) << 3;
#pragma unroll
    for (int i = 0; i < 4; ++i) {
#pragma unroll
        for (int j = 0; j < 2; ++j) {
            const int f = n0 + 16 * j + col;
            const float bv = bias[f];
#pragma unroll
            for (int r = 0; r < 8; ++r) {
                const int m = m0 + 16 * i + rhi + r;
                y[(size_t)m * CC + f] = acc[i][j][r] + bv;
            }
        }
    }
}

// ---------------------------------------------------------------------------
extern "C" void kernel_launch(void* const* d_in, const int* in_sizes, int n_in,
                              void* d_out, int out_size, void* d_ws, size_t ws_size,
                              hipStream_t stream) {
    const float* x     = (const float*)d_in[0];   // [B,T,C]
    const float* qkv_w = (const float*)d_in[1];   // [3C,C]
    const float* qkv_b = (const float*)d_in[2];   // [3C]
    const float* out_w = (const float*)d_in[3];   // [C,C]
    const float* out_b = (const float*)d_in[4];   // [C]
    float* out = (float*)d_out;                   // [B,T,C]

    const size_t NE  = (size_t)BB * HH * TT * DD; // 8,388,608 elems (16 MB f16)
    const size_t NW1 = (size_t)3 * CC * CC;       // 3,145,728
    const size_t NW2 = (size_t)CC * CC;           // 1,048,576
    _Float16* Qb = (_Float16*)d_ws;
    _Float16* Kb = Qb + NE;
    _Float16* Vt = Kb + NE;
    _Float16* Ob = Vt + NE;                       // [B,T,C] f16
    _Float16* xh = Ob + NE;                       // [B*T, C] f16
    _Float16* wqkvh = xh + NE;                    // [3C, C] f16
    _Float16* wouth = wqkvh + NW1;                // [C, C] f16

    // one-shot fp32 -> f16 staging (per-element once, not per-tile)
    cvt_f16<<<dim3((int)(NE / 8 / 256)), 256, 0, stream>>>(x, xh, (int)(NE / 8));
    cvt_f16<<<dim3((int)(NW1 / 8 / 256)), 256, 0, stream>>>(qkv_w, wqkvh, (int)(NW1 / 8));
    cvt_f16<<<dim3((int)(NW2 / 8 / 256)), 256, 0, stream>>>(out_w, wouth, (int)(NW2 / 8));

    qkv_gemm<<<dim3(3 * CC / 128, BB * TT / 128), 256, 0, stream>>>(
        xh, wqkvh, qkv_b, Qb, Kb, Vt);
    flash_attn<<<dim3(TT / 128, BB * HH), 256, 0, stream>>>(Qb, Kb, Vt, Ob);
    out_gemm<<<dim3(CC / 128, BB * TT / 128), 256, 0, stream>>>(
        Ob, wouth, out_b, out);
}
